// DirichletGaussianLayer_12240656793880
// MI455X (gfx1250) — compile-verified, matrix operand order checked by reading, not run on hardware
//
#include <hip/hip_runtime.h>
#include <hip/hip_bf16.h>
#include <math.h>

#define DIMN 256
#define KN   128
#define PITCH 264              // bf16 elems per LDS row (528 B) -> conflict-free b128 reads
#define CHALF 235.2482645f     // 0.5 * DIM * log(2*pi)

typedef __attribute__((ext_vector_type(16))) __bf16 v16bf;
typedef __attribute__((ext_vector_type(8)))  float  v8f;
typedef __attribute__((ext_vector_type(4)))  unsigned int u32x4;
typedef __attribute__((ext_vector_type(8)))  int i32x8;
typedef __attribute__((ext_vector_type(4)))  int i32x4;

__device__ __forceinline__ unsigned short bf16_rne(float f) {
  unsigned u = __builtin_bit_cast(unsigned, f);
  unsigned r = u + 0x7FFFu + ((u >> 16) & 1u);
  return (unsigned short)(r >> 16);
}
__device__ __forceinline__ float bf16_to_f(unsigned short h) {
  return __builtin_bit_cast(float, ((unsigned)h) << 16);
}

// ---------------- prep: per-component coefficients + bf16 split of mean ----
__global__ void dgl_prep(const float* __restrict__ noise_base,
                         const float* __restrict__ comp_mean,
                         const float* __restrict__ unc,
                         const float* __restrict__ lcov,
                         const float* __restrict__ rawV,
                         unsigned short* __restrict__ wsH,
                         unsigned short* __restrict__ wsL,
                         float* __restrict__ invK, float* __restrict__ offK,
                         float* __restrict__ sKv,  float* __restrict__ logV,
                         float* __restrict__ l1mV, float* __restrict__ constP) {
  __shared__ float red[KN];
  int k = threadIdx.x;
  float us = __expf(unc[k]);
  float mm = 0.f, nn = 0.f, s05 = 0.f;
  for (int d = 0; d < DIMN; ++d) {
    float nb = noise_base[k * DIMN + d] * us;
    float m  = comp_mean[k * DIMN + d] + nb;
    mm += m * m;
    nn += nb * nb;
    float dm = m - 0.5f;
    s05 += dm * dm;
    unsigned short h = bf16_rne(m);
    wsH[k * DIMN + d] = h;
    wsL[k * DIMN + d] = bf16_rne(m - bf16_to_f(h));
  }
  float var = fmaxf(__expf(lcov[k]), 1e-6f);
  float inv = 1.0f / var;
  invK[k] = inv;
  sKv[k]  = -0.5f * inv;
  offK[k] = -0.5f * ((float)DIMN * lcov[k] + mm * inv);
  float V = 1.0f / (1.0f + __expf(-rawV[k]));
  float lV = __logf(V), l1m = __logf(1.0f - V);
  logV[k] = lV;
  l1mV[k] = l1m;
  float tau = inv;
  float pms = 0.5f * __logf(tau) - tau - 0.5f * tau * s05;   // NormalGamma brief
  float pv  = (2.0f - 1.0f) * l1m;                           // Beta(1, ALPHA=2) brief
  float qd  = 0.5f * ((float)DIMN * unc[k] + nn / us);       // deterministic Q part
  red[k] = pv + pms + qd;
  __syncthreads();
  if (k == 0) {
    float s = 0.f;
    for (int i = 0; i < KN; ++i) s += red[i];   // ordered -> deterministic
    *constP = s;
  }
}

// ---------------- main: fused GEMM (split-bf16 WMMA) + softmax + loss ------
__global__ void __launch_bounds__(256) dgl_main(
    const float* __restrict__ x,
    const unsigned short* __restrict__ wsH, const unsigned short* __restrict__ wsL,
    const float* __restrict__ g_invK, const float* __restrict__ g_offK,
    const float* __restrict__ g_sK,   const float* __restrict__ g_logV,
    const float* __restrict__ g_l1mV,
    float* __restrict__ zout, float* __restrict__ partials) {
  extern __shared__ __align__(16) char smem[];
  __bf16* mH = (__bf16*)smem;                         // 128 x 264 bf16 (TDM-padded)
  __bf16* mL = (__bf16*)(smem + KN * PITCH * 2);      // 128 x 264 bf16
  float* invK = (float*)(smem + KN * PITCH * 4);
  float* offK = invK + KN;
  float* sK   = offK + KN;
  float* logV = sK + KN;
  float* l1mV = logV + KN;
  float* wred = l1mV + KN;                            // 32 floats

  int tid = threadIdx.x;
  if (tid < KN) {
    invK[tid] = g_invK[tid]; offK[tid] = g_offK[tid]; sK[tid] = g_sK[tid];
    logV[tid] = g_logV[tid]; l1mV[tid] = g_l1mV[tid];
  }

#if __has_builtin(__builtin_amdgcn_tensor_load_to_lds)
  // Tensor Data Mover: DMA the 128x256 bf16 hi/lo mean matrices into LDS.
  // D# describes a 2D tensor (dim0=256, dim1=128, 2B elems) with the whole
  // tensor as one tile; pad_enable inserts 4 DWORDs (16B) after every
  // 128 DWORDs (512B row) -> hardware produces the 528B conflict-free pitch.
  if (tid < 32) {   // wave 0 issues both descriptors (TDM ignores EXEC)
    unsigned ldsBase = (unsigned)(unsigned long long)(void*)smem;
    unsigned long long gH = (unsigned long long)wsH;
    unsigned long long gL = (unsigned long long)wsL;
    u32x4 g0H = { 1u,                                  // count=1, user descriptor
                  ldsBase,                             // lds_addr
                  (unsigned)gH,                        // global_addr[31:0]
                  (unsigned)((gH >> 32) & 0x01FFFFFFull) | (2u << 30) }; // type=2
    u32x4 g0L = { 1u,
                  ldsBase + (unsigned)(KN * PITCH * 2),
                  (unsigned)gL,
                  (unsigned)((gL >> 32) & 0x01FFFFFFull) | (2u << 30) };
    i32x8 g1 = { (int)((1u << 16)      // data_size = 1 (2-byte elems)
                     | (1u << 20)      // pad_enable
                     | (6u << 22)      // pad_interval code 6 = 128 DWORDs
                     | (3u << 25)),    // pad_amount  code 3 = 4 DWORDs
                 (int)(256u << 16),    // tensor_dim0 = 256 (bits 79:48)
                 (int)(128u << 16),    // tensor_dim1 = 128 (bits 111:80)
                 (int)(256u << 16),    // tile_dim0 = 256 (bits 127:112)
                 128,                  // tile_dim1 = 128 (bits 143:128)
                 256,                  // tensor_dim0_stride = 256 (bits 207:160)
                 0, 0 };
    i32x4 z4 = {0, 0, 0, 0};
#if defined(__clang_major__) && __clang_major__ >= 23
    i32x8 z8 = {0, 0, 0, 0, 0, 0, 0, 0};
    __builtin_amdgcn_tensor_load_to_lds(g0H, g1, z4, z4, z8, 0);
    __builtin_amdgcn_tensor_load_to_lds(g0L, g1, z4, z4, z8, 0);
#else
    __builtin_amdgcn_tensor_load_to_lds(g0H, g1, z4, z4, 0);
    __builtin_amdgcn_tensor_load_to_lds(g0L, g1, z4, z4, 0);
#endif
#if __has_builtin(__builtin_amdgcn_s_wait_tensorcnt)
    __builtin_amdgcn_s_wait_tensorcnt(0);
#endif
  }
#else
  // Fallback: manual staged copy into padded LDS (b128 granularity)
  for (int v = tid; v < (KN * DIMN) / 8; v += 256) {
    int k  = v >> 5;          // 32 vec8 per row
    int d8 = v & 31;
    uint4 h = ((const uint4*)wsH)[v];
    uint4 l = ((const uint4*)wsL)[v];
    *((uint4*)(mH + k * PITCH + d8 * 8)) = h;
    *((uint4*)(mL + k * PITCH + d8 * 8)) = l;
  }
#endif
  __syncthreads();

  int wave = tid >> 5, lane = tid & 31;
  int seg = lane & 15, half = lane >> 4;
  int rowBase = blockIdx.x * 128 + wave * 16;
  const float* xr = x + (size_t)(rowBase + seg) * DIMN;
  int c0 = half * 8;            // A/B K-chunk offsets per wave32 half
  int c1 = c0 + 16;

  v8f acc[8];
  v8f zeroA = {0.f, 0.f, 0.f, 0.f, 0.f, 0.f, 0.f, 0.f};
#pragma unroll
  for (int t = 0; t < 8; ++t) acc[t] = zeroA;
  float xx = 0.0f;

  for (int dt = 0; dt < DIMN; dt += 32) {
    if (dt + 32 < DIMN) __builtin_prefetch(xr + dt + 32 + c0, 0, 0);
    float4 p0 = *(const float4*)(xr + dt + c0);
    float4 p1 = *(const float4*)(xr + dt + c0 + 4);
    float4 q0 = *(const float4*)(xr + dt + c1);
    float4 q1 = *(const float4*)(xr + dt + c1 + 4);
    float f[16] = {p0.x, p0.y, p0.z, p0.w, p1.x, p1.y, p1.z, p1.w,
                   q0.x, q0.y, q0.z, q0.w, q1.x, q1.y, q1.z, q1.w};
    v16bf aH, aL;
#pragma unroll
    for (int i = 0; i < 16; ++i) {
      xx += f[i] * f[i];
      unsigned short h = bf16_rne(f[i]);
      aH[i] = __builtin_bit_cast(__bf16, h);
      aL[i] = __builtin_bit_cast(__bf16, bf16_rne(f[i] - bf16_to_f(h)));
    }
#pragma unroll
    for (int t = 0; t < 8; ++t) {
      int col = t * 16 + seg;
      union { uint4 u[2]; v16bf v; } BH, BL;
      BH.u[0] = *(const uint4*)(mH + col * PITCH + dt + c0);
      BH.u[1] = *(const uint4*)(mH + col * PITCH + dt + c1);
      BL.u[0] = *(const uint4*)(mL + col * PITCH + dt + c0);
      BL.u[1] = *(const uint4*)(mL + col * PITCH + dt + c1);
      // split-precision product: xH*mH + xH*mL + xL*mH
      acc[t] = __builtin_amdgcn_wmma_f32_16x16x32_bf16(false, aH, false, BH.v,
                                                       (short)0, acc[t], false, false);
      acc[t] = __builtin_amdgcn_wmma_f32_16x16x32_bf16(false, aH, false, BL.v,
                                                       (short)0, acc[t], false, false);
      acc[t] = __builtin_amdgcn_wmma_f32_16x16x32_bf16(false, aL, false, BH.v,
                                                       (short)0, acc[t], false, false);
    }
  }
  xx += __shfl_xor(xx, 16, 32);   // combine per-half partial ||x||^2

  float a1s = 0.f, a2s = 0.f, a3s = 0.f, a4s = 0.f;
#pragma unroll
  for (int r = 0; r < 8; ++r) {
    int rowM = rowBase + half * 8 + r;            // row owned by this half
    float xxr = __shfl(xx, half * 8 + r, 32);
    float lg[8], ee[8], pre[8];
    float mx = -3.0e38f;
#pragma unroll
    for (int t = 0; t < 8; ++t) {
      int k = seg + 16 * t;
      lg[t] = sK[k] * xxr + invK[k] * acc[t][r] + offK[k];
      mx = fmaxf(mx, lg[t]);
    }
    mx = fmaxf(mx, __shfl_xor(mx, 1, 32));
    mx = fmaxf(mx, __shfl_xor(mx, 2, 32));
    mx = fmaxf(mx, __shfl_xor(mx, 4, 32));
    mx = fmaxf(mx, __shfl_xor(mx, 8, 32));
    // exp + 16-lane inclusive prefix (for stick-breaking cumsum) + total
    float running = 0.0f;
#pragma unroll
    for (int t = 0; t < 8; ++t) {
      float e = __expf(lg[t] - mx);
      ee[t] = e;
      float p = e;
#pragma unroll
      for (int d = 1; d <= 8; d <<= 1) {
        float u = __shfl_up(p, d, 16);
        if (seg >= d) p += u;
      }
      float tot = __shfl(p, 15, 16);
      pre[t] = running + p;
      running += tot;
    }
    float inv = 1.0f / running;
#pragma unroll
    for (int t = 0; t < 8; ++t) {
      int k = seg + 16 * t;
      float z = ee[t] * inv;
      a1s += z * logV[k];
      a2s += z * (lg[t] - CHALF);
      a3s += z * __logf(fmaxf(z, 1e-6f));
      float cum = fminf(pre[t] * inv, 1.0f);
      if (k <= KN - 2) a4s += (1.0f - cum) * l1mV[k];
      zout[(size_t)rowM * KN + k] = z;
    }
  }

  // wave reduction, then ordered block reduction -> per-block partials
#pragma unroll
  for (int d = 1; d < 32; d <<= 1) {
    a1s += __shfl_xor(a1s, d, 32);
    a2s += __shfl_xor(a2s, d, 32);
    a3s += __shfl_xor(a3s, d, 32);
    a4s += __shfl_xor(a4s, d, 32);
  }
  if (lane == 0) {
    wred[wave * 4 + 0] = a1s; wred[wave * 4 + 1] = a2s;
    wred[wave * 4 + 2] = a3s; wred[wave * 4 + 3] = a4s;
  }
  __syncthreads();
  if (tid < 4) {
    float s = 0.f;
    for (int w = 0; w < 8; ++w) s += wred[w * 4 + tid];   // ordered
    partials[(size_t)tid * gridDim.x + blockIdx.x] = s;
  }
}

// ---------------- finalize: deterministic ordered sum -> scalar -----------
__global__ void dgl_finalize(const float* __restrict__ partials,
                             const float* __restrict__ constP,
                             float* __restrict__ out, int nblocks, float invN) {
  if (threadIdx.x == 0 && blockIdx.x == 0) {
    float S[4];
    for (int c = 0; c < 4; ++c) {
      float s = 0.f;
      for (int b = 0; b < nblocks; ++b) s += partials[(size_t)c * nblocks + b];
      S[c] = s;
    }
    // LOBO = constP + (64*(A_zlogV + A_px + A_cum) - A_entropy)/N
    float LOBO = *constP + (64.0f * (S[0] + S[1] + S[3]) - S[2]) * invN;
    out[0] = -LOBO;
  }
}

extern "C" void kernel_launch(void* const* d_in, const int* in_sizes, int n_in,
                              void* d_out, int out_size, void* d_ws, size_t ws_size,
                              hipStream_t stream) {
  (void)n_in; (void)out_size; (void)ws_size;
  const float* x          = (const float*)d_in[0];
  const float* noise_base = (const float*)d_in[1];
  const float* comp_mean  = (const float*)d_in[2];
  const float* unc        = (const float*)d_in[3];
  const float* lcov       = (const float*)d_in[4];
  const float* rawV       = (const float*)d_in[5];
  int N = in_sizes[0] / DIMN;
  int nblocks = N / 128;

  char* ws = (char*)d_ws;
  unsigned short* wsH = (unsigned short*)ws;                        // K*D bf16
  unsigned short* wsL = (unsigned short*)(ws + (size_t)KN * DIMN * 2);
  float* invK   = (float*)(ws + (size_t)KN * DIMN * 4);
  float* offK   = invK + KN;
  float* sKv    = offK + KN;
  float* logV   = sKv + KN;
  float* l1mV   = logV + KN;
  float* constP = l1mV + KN;
  float* partials = constP + 16;                                    // 4*nblocks floats
  float* out = (float*)d_out;

  size_t smem = (size_t)KN * PITCH * 4 + 5 * KN * 4 + 32 * 4;       // ~137.9 KB
  hipFuncSetAttribute((const void*)dgl_main,
                      hipFuncAttributeMaxDynamicSharedMemorySize, (int)smem);

  dgl_prep<<<1, KN, 0, stream>>>(noise_base, comp_mean, unc, lcov, rawV,
                                 wsH, wsL, invK, offK, sKv, logV, l1mV, constP);
  dgl_main<<<nblocks, 256, smem, stream>>>(x, wsH, wsL, invK, offK, sKv, logV,
                                           l1mV, out + 1, partials);
  dgl_finalize<<<1, 1, 0, stream>>>(partials, constP, out, nblocks,
                                    1.0f / (float)N);
}
